// Classifier_69818988363910
// MI455X (gfx1250) — matrix operand order repaired and checked by reading, not compiled, not run on hardware
//
#include <hip/hip_runtime.h>
#include <hip/hip_bf16.h>

// ---------------------------------------------------------------------------
// Classifier: logits[2000,1000] = mean_J(relu(x @ W1^T)) @ (WlinL + WlinR)^T
//   T=60000, D=1024, E=2048, C=1000, J=30, S=T/J=2000
// y/W2 branch dead (reference bug) -> skipped.
// concat(features,features) @ Wlin^T == features @ (Wlin[:,:E]+Wlin[:,E:])^T
// Heavy GEMM (252 GFLOP) on v_wmma_f32_16x16x32_bf16, f32 accumulate.
// B tiles: global->LDS via async DMA (global_load_async_to_lds_b128, no VGPR
// staging, ASYNCcnt-tracked), double-buffered LDS. A tiles: f32->bf16 via
// v_cvt_pk_bf16_f32 on the VGPR path (8..16 regs only).
// Workspace: W1bf16 (4 MB) | Wsumbf16 (4 MB) | features f32 (16 MB) ~= 24 MB.
// ---------------------------------------------------------------------------

typedef __attribute__((ext_vector_type(16))) __bf16 v16bf_t;
typedef __attribute__((ext_vector_type(8)))  float  v8f_t;

union Frag {
    v16bf_t v;
    uint4   q[2];
};

union BfPack8 {            // 8 f32 -> 8 bf16 (hardware v_cvt_pk_bf16_f32)
    __bf16 h[8];
    uint4  q;
};

union BfPack4 {
    __bf16 h[4];
    uint2  q;
};

__device__ __forceinline__ void wait_asynccnt0() {
#if __has_builtin(__builtin_amdgcn_s_wait_asynccnt)
    __builtin_amdgcn_s_wait_asynccnt(0);
#else
    asm volatile("s_wait_asynccnt 0x0" ::: "memory");
#endif
}

// async DMA: 16 bytes global -> LDS, no VGPR data path (ASYNCcnt)
__device__ __forceinline__ void async_b128(unsigned lds_addr,
                                           const unsigned short* g) {
    asm volatile("global_load_async_to_lds_b128 %0, %1, off"
                 :: "v"(lds_addr), "v"(g)
                 : "memory");
}

// ----------------------------- prep kernels --------------------------------

__global__ void cvt_w1_kernel(const float* __restrict__ W1,
                              unsigned int* __restrict__ W1bf) {
    int i = blockIdx.x * blockDim.x + threadIdx.x;   // (2048*1024)/4 threads
    float4 v = ((const float4*)W1)[i];
    BfPack4 p;
    p.h[0] = (__bf16)v.x; p.h[1] = (__bf16)v.y;
    p.h[2] = (__bf16)v.z; p.h[3] = (__bf16)v.w;
    ((uint2*)W1bf)[i] = p.q;
}

__global__ void make_wsum_kernel(const float* __restrict__ Wlin,
                                 unsigned int* __restrict__ Wsum) {
    int i  = blockIdx.x * blockDim.x + threadIdx.x;  // (1000*2048)/4 threads
    int i4 = i << 2;
    int c  = i4 >> 11;           // row in [0,1000)
    int f  = i4 & 2047;          // col in [0,2048), multiple of 4
    float4 a = *(const float4*)(Wlin + (size_t)c * 4096 + f);
    float4 b = *(const float4*)(Wlin + (size_t)c * 4096 + 2048 + f);
    BfPack4 p;
    p.h[0] = (__bf16)(a.x + b.x); p.h[1] = (__bf16)(a.y + b.y);
    p.h[2] = (__bf16)(a.z + b.z); p.h[3] = (__bf16)(a.w + b.w);
    ((uint2*)Wsum)[i] = p.q;
}

// --------------------------- GEMM tiling params ----------------------------
// BM=32, BN=128, BK=64; 128 threads = 4 waves. Wave w owns cols [w*32,w*32+32)
// as two 16-col tiles x two 16-row tiles (4 accumulators, 2 ds_load per WMMA).
// Double-buffered LDS; row stride 72 u16 = 144 B (16B aligned, staggered).

#define LDA 72
#define LDB 72

// GEMM1: A = x[seg*30..seg*30+29][1024] (f32->bf16 on the fly, rows 30/31
// zero), B = W1bf[2048][1024] via async DMA. Epilogue: relu + mean/30.
__global__ __launch_bounds__(128)
void gemm1_relu_mean(const float* __restrict__ x,
                     const unsigned short* __restrict__ Bw,
                     float* __restrict__ feat) {
    __shared__ unsigned short As[2][32 * LDA];
    __shared__ unsigned short Bs[2][128 * LDB];

    const int seg   = blockIdx.y;          // 0..1999
    const int nbase = blockIdx.x * 128;    // 0..1920
    const int t     = threadIdx.x;
    const int lane  = t & 31;
    const int wv    = t >> 5;              // 0..3

    // staging: A 32x64 f32 (16/thread); B row t via 8 async b128
    const int ar = t >> 2;                 // 0..31
    const int ac = (t & 3) << 4;           // 0,16,32,48
    const bool arow = ar < 30;
    const float*          xa = x  + (size_t)(seg * 30 + ar) * 1024 + ac;
    const unsigned short* bg = Bw + (size_t)(nbase + t) * 1024;

    v8f_t acc00 = {}, acc01 = {}, acc10 = {}, acc11 = {};
    float areg[16];

    // fragment addressing (CDNA5 VGPR layouts)
    const int am  = lane & 15;
    const int ak  = (lane >> 4) << 3;      // A: lanes>=16 hold K+8 block
    const int bn0 = (wv << 5) + (lane & 15);
    const int bk  = (lane >> 4) << 3;      // B: lanes>=16 hold K+8 block

    auto async_B = [&](int buf, int k0) {
        const unsigned short* g = bg + k0;
        unsigned l = (unsigned)(uintptr_t)&Bs[buf][t * LDB];
#pragma unroll
        for (int j = 0; j < 8; ++j)
            async_b128(l + j * 16, g + j * 8);
    };

    auto load_A = [&](int k0) {
        if (arow) {
            const float4* p = (const float4*)(xa + k0);
#pragma unroll
            for (int j = 0; j < 4; ++j) {
                float4 v = p[j];
                areg[4 * j + 0] = v.x; areg[4 * j + 1] = v.y;
                areg[4 * j + 2] = v.z; areg[4 * j + 3] = v.w;
            }
        } else {
#pragma unroll
            for (int j = 0; j < 16; ++j) areg[j] = 0.f;
        }
    };

    auto store_A = [&](int buf) {
        BfPack8 p0, p1;
#pragma unroll
        for (int j = 0; j < 8; ++j) {
            p0.h[j] = (__bf16)areg[j];
            p1.h[j] = (__bf16)areg[8 + j];
        }
        uint4* d = (uint4*)&As[buf][ar * LDA + ac];
        d[0] = p0.q;
        d[1] = p1.q;
    };

    auto compute = [&](int buf) {
#pragma unroll
        for (int ks = 0; ks < 2; ++ks) {
            Frag a0, a1, b0, b1;
            const unsigned short* ap0 = &As[buf][am * LDA + ak + ks * 32];
            a0.q[0] = *(const uint4*)ap0;
            a0.q[1] = *(const uint4*)(ap0 + 16);
            const unsigned short* ap1 = &As[buf][(am + 16) * LDA + ak + ks * 32];
            a1.q[0] = *(const uint4*)ap1;
            a1.q[1] = *(const uint4*)(ap1 + 16);
            const unsigned short* bp0 = &Bs[buf][bn0 * LDB + bk + ks * 32];
            b0.q[0] = *(const uint4*)bp0;
            b0.q[1] = *(const uint4*)(bp0 + 16);
            const unsigned short* bp1 = &Bs[buf][(bn0 + 16) * LDB + bk + ks * 32];
            b1.q[0] = *(const uint4*)bp1;
            b1.q[1] = *(const uint4*)(bp1 + 16);
            acc00 = __builtin_amdgcn_wmma_f32_16x16x32_bf16(
                false, a0.v, false, b0.v, (short)0, acc00, false, false);
            acc10 = __builtin_amdgcn_wmma_f32_16x16x32_bf16(
                false, a1.v, false, b0.v, (short)0, acc10, false, false);
            acc01 = __builtin_amdgcn_wmma_f32_16x16x32_bf16(
                false, a0.v, false, b1.v, (short)0, acc01, false, false);
            acc11 = __builtin_amdgcn_wmma_f32_16x16x32_bf16(
                false, a1.v, false, b1.v, (short)0, acc11, false, false);
        }
    };

    // prologue: fill buffer 0
    async_B(0, 0);
    load_A(0);
    store_A(0);
    wait_asynccnt0();
    __syncthreads();

#pragma unroll 1
    for (int c = 0; c < 16; ++c) {         // K = 1024 / 64
        const int cb = c & 1, nb = cb ^ 1;
        if (c + 1 < 16) {
            async_B(nb, (c + 1) * 64);     // DMA next B tile into other buffer
            load_A((c + 1) * 64);          // prefetch next A tile to regs
        }
        compute(cb);                       // WMMAs hide DMA + A-load latency
        if (c + 1 < 16) {
            store_A(nb);
            wait_asynccnt0();
            __syncthreads();
        }
    }

    // epilogue: relu + mean over 30 rows (rows 30/31 are zero-padded)
    float ps0 = 0.f, ps1 = 0.f;
#pragma unroll
    for (int i = 0; i < 8; ++i) {
        ps0 += fmaxf(acc00[i], 0.f) + fmaxf(acc10[i], 0.f);
        ps1 += fmaxf(acc01[i], 0.f) + fmaxf(acc11[i], 0.f);
    }
    ps0 += __shfl_xor(ps0, 16, 32);        // merge row-halves per column
    ps1 += __shfl_xor(ps1, 16, 32);
    if (lane < 16) {
        float* fp = feat + (size_t)seg * 2048 + nbase + (wv << 5) + lane;
        fp[0]  = ps0 * (1.0f / 30.0f);
        fp[16] = ps1 * (1.0f / 30.0f);
    }
}

// GEMM2: out[2000,1000] = feat[2000,2048] @ Wsum^T; row/col guards for the
// 2016x1024 padded grid; invalid B rows pre-zeroed in LDS (async skipped).
__global__ __launch_bounds__(128)
void gemm2_logits(const float* __restrict__ feat,
                  const unsigned short* __restrict__ Bw,
                  float* __restrict__ out) {
    __shared__ unsigned short As[2][32 * LDA];
    __shared__ unsigned short Bs[2][128 * LDB];

    const int mbase = blockIdx.y * 32;     // 0..1984
    const int nbase = blockIdx.x * 128;    // 0..896
    const int t     = threadIdx.x;
    const int lane  = t & 31;
    const int wv    = t >> 5;

    const int ar = t >> 2;
    const int ac = (t & 3) << 4;
    const bool avalid = (mbase + ar) < 2000;
    const bool bvalid = (nbase + t)  < 1000;
    const float*          xa = feat + (size_t)(mbase + ar) * 2048 + ac;
    const unsigned short* bg = Bw   + (size_t)(nbase + t) * 2048;

    v8f_t acc00 = {}, acc01 = {}, acc10 = {}, acc11 = {};
    float areg[16];

    const int am  = lane & 15;
    const int ak  = (lane >> 4) << 3;
    const int bn0 = (wv << 5) + (lane & 15);
    const int bk  = (lane >> 4) << 3;

    auto async_B = [&](int buf, int k0) {
        if (bvalid) {
            const unsigned short* g = bg + k0;
            unsigned l = (unsigned)(uintptr_t)&Bs[buf][t * LDB];
#pragma unroll
            for (int j = 0; j < 8; ++j)
                async_b128(l + j * 16, g + j * 8);
        }
    };

    auto load_A = [&](int k0) {
        if (avalid) {
            const float4* p = (const float4*)(xa + k0);
#pragma unroll
            for (int j = 0; j < 4; ++j) {
                float4 v = p[j];
                areg[4 * j + 0] = v.x; areg[4 * j + 1] = v.y;
                areg[4 * j + 2] = v.z; areg[4 * j + 3] = v.w;
            }
        } else {
#pragma unroll
            for (int j = 0; j < 16; ++j) areg[j] = 0.f;
        }
    };

    auto store_A = [&](int buf) {
        BfPack8 p0, p1;
#pragma unroll
        for (int j = 0; j < 8; ++j) {
            p0.h[j] = (__bf16)areg[j];
            p1.h[j] = (__bf16)areg[8 + j];
        }
        uint4* d = (uint4*)&As[buf][ar * LDA + ac];
        d[0] = p0.q;
        d[1] = p1.q;
    };

    auto compute = [&](int buf) {
#pragma unroll
        for (int ks = 0; ks < 2; ++ks) {
            Frag a0, a1, b0, b1;
            const unsigned short* ap0 = &As[buf][am * LDA + ak + ks * 32];
            a0.q[0] = *(const uint4*)ap0;
            a0.q[1] = *(const uint4*)(ap0 + 16);
            const unsigned short* ap1 = &As[buf][(am + 16) * LDA + ak + ks * 32];
            a1.q[0] = *(const uint4*)ap1;
            a1.q[1] = *(const uint4*)(ap1 + 16);
            const unsigned short* bp0 = &Bs[buf][bn0 * LDB + bk + ks * 32];
            b0.q[0] = *(const uint4*)bp0;
            b0.q[1] = *(const uint4*)(bp0 + 16);
            const unsigned short* bp1 = &Bs[buf][(bn0 + 16) * LDB + bk + ks * 32];
            b1.q[0] = *(const uint4*)bp1;
            b1.q[1] = *(const uint4*)(bp1 + 16);
            acc00 = __builtin_amdgcn_wmma_f32_16x16x32_bf16(
                false, a0.v, false, b0.v, (short)0, acc00, false, false);
            acc10 = __builtin_amdgcn_wmma_f32_16x16x32_bf16(
                false, a1.v, false, b0.v, (short)0, acc10, false, false);
            acc01 = __builtin_amdgcn_wmma_f32_16x16x32_bf16(
                false, a0.v, false, b1.v, (short)0, acc01, false, false);
            acc11 = __builtin_amdgcn_wmma_f32_16x16x32_bf16(
                false, a1.v, false, b1.v, (short)0, acc11, false, false);
        }
    };

    // pre-zero LDS rows whose B row is out of range (stay zero all chunks)
    if (!bvalid) {
        uint4 z = make_uint4(0u, 0u, 0u, 0u);
        uint4* b0 = (uint4*)&Bs[0][t * LDB];
        uint4* b1 = (uint4*)&Bs[1][t * LDB];
#pragma unroll
        for (int j = 0; j < 8; ++j) { b0[j] = z; b1[j] = z; }
    }

    async_B(0, 0);
    load_A(0);
    store_A(0);
    wait_asynccnt0();
    __syncthreads();

#pragma unroll 1
    for (int c = 0; c < 32; ++c) {         // K = 2048 / 64
        const int cb = c & 1, nb = cb ^ 1;
        if (c + 1 < 32) {
            async_B(nb, (c + 1) * 64);
            load_A((c + 1) * 64);
        }
        compute(cb);
        if (c + 1 < 32) {
            store_A(nb);
            wait_asynccnt0();
            __syncthreads();
        }
    }

    const int col0 = nbase + (wv << 5) + (lane & 15);
    const int col1 = col0 + 16;
    const int r0   = mbase + ((lane >> 4) << 3);
    if (col0 < 1000) {
#pragma unroll
        for (int i = 0; i < 8; ++i) {
            int ra = r0 + i;
            if (ra < 2000) out[(size_t)ra * 1000 + col0] = acc00[i];
            int rb = r0 + 16 + i;
            if (rb < 2000) out[(size_t)rb * 1000 + col0] = acc10[i];
        }
    }
    if (col1 < 1000) {
#pragma unroll
        for (int i = 0; i < 8; ++i) {
            int ra = r0 + i;
            if (ra < 2000) out[(size_t)ra * 1000 + col1] = acc01[i];
            int rb = r0 + 16 + i;
            if (rb < 2000) out[(size_t)rb * 1000 + col1] = acc11[i];
        }
    }
}

// ------------------------------- launcher ----------------------------------

extern "C" void kernel_launch(void* const* d_in, const int* in_sizes, int n_in,
                              void* d_out, int out_size, void* d_ws, size_t ws_size,
                              hipStream_t stream) {
    const float* x    = (const float*)d_in[0];   // [60000,1024]
    // d_in[1] = y   : dead branch, skipped
    const float* W1   = (const float*)d_in[2];   // [2048,1024]
    // d_in[3] = W2  : dead branch, skipped
    const float* Wlin = (const float*)d_in[4];   // [1000,4096]
    float* out = (float*)d_out;                  // [2000,1000]

    char* ws = (char*)d_ws;
    unsigned short* W1bf   = (unsigned short*)(ws);                 // 4,194,304 B
    unsigned short* Wsumbf = (unsigned short*)(ws + 4194304);       // 4,096,000 B
    float*          feat   = (float*)(ws + 8290304);                // 16,384,000 B
    // total workspace: ~24.7 MB

    cvt_w1_kernel   <<<2048, 256, 0, stream>>>(W1,   (unsigned int*)W1bf);   // 2048*1024/4
    make_wsum_kernel<<<2000, 256, 0, stream>>>(Wlin, (unsigned int*)Wsumbf); // 1000*2048/4

    gemm1_relu_mean<<<dim3(16, 2000), 128, 0, stream>>>(x, W1bf, feat);
    gemm2_logits   <<<dim3(8,  63),  128, 0, stream>>>(feat, Wsumbf, out);
}
